// SwinTransformerBlock_36326833389754
// MI455X (gfx1250) — matrix operand order repaired and checked by reading, not compiled, not run on hardware
//
#include <hip/hip_runtime.h>
#include <hip/hip_bf16.h>
#include <math.h>

// ---------------- problem constants ----------------
#define Bq   8
#define Sq   4096
#define Cq   512
#define Hq   16
#define Wq   64
#define SHIFTq 32
#define HDq  32          // C/H
#define NWq  64          // S/W
#define ROWS (Bq*Sq)     // 32768 tokens

typedef __attribute__((ext_vector_type(16))) _Float16 v16h;
typedef __attribute__((ext_vector_type(8)))  float    v8f;

// ---------------- CDNA5 async global->LDS copy path ----------------
#if defined(__HIP_DEVICE_COMPILE__) && __has_builtin(__builtin_amdgcn_global_load_async_to_lds_b128)
#define USE_ASYNC 1
#endif

#ifdef USE_ASYNC
// builtin expects: (int4 __global*, int4 __shared*, imm offset, imm cpol)
typedef int v4i __attribute__((__vector_size__(16)));
typedef __attribute__((address_space(1))) v4i gv4i;
typedef __attribute__((address_space(3))) v4i lv4i;
#if __has_builtin(__builtin_amdgcn_s_wait_asynccnt)
#define WAIT_ASYNC(n) __builtin_amdgcn_s_wait_asynccnt(n)
#else
#define WAIT_ASYNC(n) asm volatile("s_wait_asynccnt %0" :: "i"(n) : "memory")
#endif
#else
#define WAIT_ASYNC(n) ((void)0)
#endif

// copy 16 bytes global -> LDS (async on CDNA5, sync fallback otherwise)
__device__ __forceinline__ void copy16(const _Float16* g, _Float16* l) {
#ifdef USE_ASYNC
  __builtin_amdgcn_global_load_async_to_lds_b128((gv4i*)g, (lv4i*)l, 0, 0);
#else
  *(uint4*)l = *(const uint4*)g;
#endif
}

// stage one 128-row x 32-col fp16 tile (8 KB) into LDS; 256 threads, 2 chunks each
__device__ __forceinline__ void stage_tile(const _Float16* __restrict__ g, int ld,
                                           _Float16* __restrict__ l, int tid) {
#pragma unroll
  for (int c = tid; c < 512; c += 256) {
    copy16(g + (size_t)(c >> 2) * ld + (c & 3) * 8, l + c * 8);
  }
}

// ---------------- WMMA helpers (gfx1250, wave32) ----------------
__device__ __forceinline__ v8f wmma_f16(v16h a, v16h b, v8f c) {
  return __builtin_amdgcn_wmma_f32_16x16x32_f16(
      /*neg_a=*/false, a, /*neg_b=*/false, b,
      /*c_mod=*/(short)0, c, /*reuse_a=*/false, /*reuse_b=*/false);
}

// A fragment: 16x32 (M x K) row-major, leading dim lda.
// lanes 0-15 -> M=lane, K {0..7,16..23}; lanes 16-31 -> K {8..15,24..31}
__device__ __forceinline__ v16h load_a_frag(const _Float16* __restrict__ A,
                                            int lda, int lane) {
  const int m     = lane & 15;
  const int khalf = (lane >> 4) & 1;
  const _Float16* p = A + (size_t)m * lda + khalf * 8;
  v16h a;
#pragma unroll
  for (int j = 0; j < 8; ++j) a[j]     = p[j];
#pragma unroll
  for (int j = 0; j < 8; ++j) a[8 + j] = p[16 + j];
  return a;
}

// B fragment: 32x16 (K x N), B[k][n] = W[n][k], row-major W with ld = ldb.
// lanes 0-15 -> N=lane, K {0..7,16..23}; lanes 16-31 -> N=lane-16, K {8..15,24..31}
__device__ __forceinline__ v16h load_b_frag(const _Float16* __restrict__ Wr,
                                            int ldb, int lane) {
  const int n    = lane & 15;
  const int ksel = (lane >> 4) & 1;
  const _Float16* p = Wr + (size_t)n * ldb + ksel * 8;
  v16h b;
#pragma unroll
  for (int j = 0; j < 8; ++j) b[j]     = p[j];
#pragma unroll
  for (int j = 0; j < 8; ++j) b[8 + j] = p[16 + j];
  return b;
}

// ---------------- LDS-staged block GEMM --------------------------------
// Block = 256 threads (8 waves), output tile 128(M) x 128(N).
// Per K-step (32): stage A(128x32) + B(128x32) tiles into LDS with async
// copies, double-buffered; each wave computes a 32x64 sub-tile (8 WMMAs).
__device__ __forceinline__ void block_gemm_lds(const _Float16* __restrict__ A, int lda,
                                               const _Float16* __restrict__ Wr, int ldb,
                                               int K,
                                               _Float16* __restrict__ Asm,
                                               _Float16* __restrict__ Bsm,
                                               int tid, v8f acc[2][4]) {
  const int lane = tid & 31;
  const int mrow = ((tid >> 5) & 3) * 32;   // wave's M offset in block tile
  const int ncol = ((tid >> 5) >> 2) * 64;  // wave's N offset in block tile

  stage_tile(A, lda, Asm, tid);             // k0 = 0 into buffer 0
  stage_tile(Wr, ldb, Bsm, tid);

  int cur = 0;
  for (int k0 = 0; k0 < K; k0 += 32) {
    const bool more = (k0 + 32) < K;
    if (more) {                             // prefetch next K-slab into other buffer
      stage_tile(A + k0 + 32, lda, Asm + (cur ^ 1) * 4096, tid);
      stage_tile(Wr + k0 + 32, ldb, Bsm + (cur ^ 1) * 4096, tid);
      WAIT_ASYNC(4);                        // allow 4 in flight, current slab done
    } else {
      WAIT_ASYNC(0);
    }
    __syncthreads();                        // all waves' copies visible

    const _Float16* At = Asm + cur * 4096;
    const _Float16* Bt = Bsm + cur * 4096;
    v16h a0 = load_a_frag(At + (size_t)mrow * 32, 32, lane);
    v16h a1 = load_a_frag(At + (size_t)(mrow + 16) * 32, 32, lane);
#pragma unroll
    for (int nt = 0; nt < 4; ++nt) {
      v16h b = load_b_frag(Bt + (size_t)(ncol + nt * 16) * 32, 32, lane);
      acc[0][nt] = wmma_f16(a0, b, acc[0][nt]);
      acc[1][nt] = wmma_f16(a1, b, acc[1][nt]);
    }
    __syncthreads();                        // done reading before refill
    cur ^= 1;
  }
}

// ---------------- fp32 -> fp16 weight conversion ----------------
__global__ void f32_to_f16_kernel(const float* __restrict__ in,
                                  _Float16* __restrict__ out, int n) {
  int i = blockIdx.x * blockDim.x + threadIdx.x;
  if (i < n) out[i] = (_Float16)in[i];
}

// ---------------- LayerNorm (+optional roll) -> fp16 ----------------
__global__ __launch_bounds__(256) void ln_kernel(const float* __restrict__ x,
                                                 const float* __restrict__ g,
                                                 const float* __restrict__ bb,
                                                 _Float16* __restrict__ out,
                                                 int shift) {
  const int wid  = (blockIdx.x * blockDim.x + threadIdx.x) >> 5;
  const int lane = threadIdx.x & 31;
  const float* row = x + (size_t)wid * Cq;

  float vals[16];
  float s = 0.f;
#pragma unroll
  for (int j = 0; j < 16; ++j) { vals[j] = row[j * 32 + lane]; s += vals[j]; }
#pragma unroll
  for (int off = 16; off > 0; off >>= 1) s += __shfl_xor(s, off, 32);
  const float mean = s * (1.0f / Cq);

  float vs = 0.f;
#pragma unroll
  for (int j = 0; j < 16; ++j) { float d = vals[j] - mean; vs += d * d; }
#pragma unroll
  for (int off = 16; off > 0; off >>= 1) vs += __shfl_xor(vs, off, 32);
  const float inv = rsqrtf(vs * (1.0f / Cq) + 1e-5f);

  const int b = wid >> 12;
  const int p = wid & (Sq - 1);
  const int dst = (p - shift) & (Sq - 1);
  _Float16* orow = out + ((size_t)(b << 12) + dst) * Cq;
#pragma unroll
  for (int j = 0; j < 16; ++j) {
    int c = j * 32 + lane;
    orow[c] = (_Float16)((vals[j] - mean) * inv * g[c] + bb[c]);
  }
}

// ---------------- QKV GEMM: (32768 x 1536 x 512) ----------------
__global__ __launch_bounds__(256) void qkv_kernel(const _Float16* __restrict__ hN,
                                                  const _Float16* __restrict__ Wqkv,
                                                  const float* __restrict__ bias,
                                                  _Float16* __restrict__ q16,
                                                  _Float16* __restrict__ k16,
                                                  _Float16* __restrict__ v16) {
  __shared__ alignas(16) _Float16 Asm[2 * 4096];
  __shared__ alignas(16) _Float16 Bsm[2 * 4096];
  const int tid = threadIdx.x;
  const int lane = tid & 31;
  const int brow0 = blockIdx.x * 128;
  const int bcol0 = blockIdx.y * 128;

  v8f acc[2][4] = {};
  block_gemm_lds(hN + (size_t)brow0 * Cq, Cq, Wqkv + (size_t)bcol0 * Cq, Cq, Cq,
                 Asm, Bsm, tid, acc);

  const int mrow = ((tid >> 5) & 3) * 32;
  const int ncol = ((tid >> 5) >> 2) * 64;
  const int cb = lane & 15;
  const int rb = (lane >> 4) << 3;
#pragma unroll
  for (int rt = 0; rt < 2; ++rt)
#pragma unroll
    for (int nt = 0; nt < 4; ++nt)
#pragma unroll
      for (int r = 0; r < 8; ++r) {
        const int row = brow0 + mrow + rt * 16 + rb + r;   // token (rolled domain)
        const int col = bcol0 + ncol + nt * 16 + cb;       // 0..1535
        float val = acc[rt][nt][r] + bias[col];
        const int kind = col >> 9;                         // 0=q 1=k 2=v
        const int cn = col & 511;
        const int hh = cn >> 5, d = cn & 31;
        const int win = row >> 6, t = row & 63;
        if (kind == 0) {
          q16[(((size_t)win * Hq + hh) * Wq + t) * HDq + d] =
              (_Float16)(val * 0.17677669529663687f);       // HD^-0.5
        } else if (kind == 1) {
          k16[(((size_t)win * Hq + hh) * Wq + t) * HDq + d] = (_Float16)val;
        } else {
          v16[(((size_t)win * Hq + hh) * HDq + d) * Wq + t] = (_Float16)val; // V^T
        }
      }
}

// ---------------- attention: one (window, head) per 128-thread block --------
__global__ __launch_bounds__(128) void attn_kernel(const _Float16* __restrict__ q16,
                                                   const _Float16* __restrict__ k16,
                                                   const _Float16* __restrict__ vT,
                                                   const float* __restrict__ rel_bias,
                                                   _Float16* __restrict__ o16) {
  __shared__ float    Sf[Wq * Wq];      // 16 KB scores
  __shared__ _Float16 Ph[Wq * Wq];      // 8 KB probs

  const int bx  = blockIdx.x;           // win*H + h
  const int win = bx >> 4;
  const int h   = bx & 15;
  const int lane = threadIdx.x & 31;
  const int wv   = threadIdx.x >> 5;
  const int m0   = wv * 16;

  const size_t base = (size_t)bx * (Wq * HDq);

  // S = (Q*scale) @ K^T (K-dim 32 -> one WMMA step)
  {
    v16h aq = load_a_frag(q16 + base + (size_t)m0 * HDq, HDq, lane);
    const int cb = lane & 15;
    const int rb = (lane >> 4) << 3;
#pragma unroll
    for (int nt = 0; nt < 4; ++nt) {
      v16h bk = load_b_frag(k16 + base + (size_t)(nt * 16) * HDq, HDq, lane);
      v8f sacc = {};
      sacc = wmma_f16(aq, bk, sacc);
#pragma unroll
      for (int r = 0; r < 8; ++r)
        Sf[(m0 + rb + r) * Wq + nt * 16 + cb] = sacc[r];
    }
  }
  __syncthreads();

  // bias + shift mask + softmax (2 threads per row)
  {
    const int t = threadIdx.x;
    const int r  = t >> 1;
    const int c0 = (t & 1) * 32;
    const bool lastwin = ((win & (NWq - 1)) == (NWq - 1));
    float buf[32];
    float mx = -1e30f;
#pragma unroll
    for (int j = 0; j < 32; ++j) {
      const int c = c0 + j;
      float v = Sf[r * Wq + c] + rel_bias[(r - c + Wq - 1) * Hq + h];
      if (lastwin && ((r < 32) != (c < 32))) v -= 100.0f;
      buf[j] = v;
      mx = fmaxf(mx, v);
    }
    mx = fmaxf(mx, __shfl_xor(mx, 1, 32));
    float sum = 0.f;
#pragma unroll
    for (int j = 0; j < 32; ++j) { float e = __expf(buf[j] - mx); buf[j] = e; sum += e; }
    sum += __shfl_xor(sum, 1, 32);
    const float inv = 1.0f / sum;
#pragma unroll
    for (int j = 0; j < 32; ++j) Ph[r * Wq + c0 + j] = (_Float16)(buf[j] * inv);
  }
  __syncthreads();

  // O = P @ V (K-dim 64 -> two WMMA steps; V pre-transposed)
  {
    const size_t vbase = (size_t)bx * (HDq * Wq);
    v8f oacc[2] = {};
#pragma unroll
    for (int k0 = 0; k0 < Wq; k0 += 32) {
      v16h ap = load_a_frag(&Ph[m0 * Wq + k0], Wq, lane);
#pragma unroll
      for (int nt = 0; nt < 2; ++nt) {
        v16h bv = load_b_frag(vT + vbase + (size_t)(nt * 16) * Wq + k0, Wq, lane);
        oacc[nt] = wmma_f16(ap, bv, oacc[nt]);
      }
    }
    const int cb = lane & 15;
    const int rb = (lane >> 4) << 3;
#pragma unroll
    for (int nt = 0; nt < 2; ++nt)
#pragma unroll
      for (int r = 0; r < 8; ++r) {
        const int row = m0 + rb + r;
        o16[((size_t)(win * Wq + row)) * Cq + h * HDq + nt * 16 + cb] =
            (_Float16)oacc[nt][r];
      }
  }
}

// ---------------- proj GEMM + roll(+SHIFT) + residual -> x1 ----------
__global__ __launch_bounds__(256) void proj_kernel(const _Float16* __restrict__ o16,
                                                   const _Float16* __restrict__ Wp,
                                                   const float* __restrict__ bias,
                                                   const float* __restrict__ x,
                                                   float* __restrict__ x1) {
  __shared__ alignas(16) _Float16 Asm[2 * 4096];
  __shared__ alignas(16) _Float16 Bsm[2 * 4096];
  const int tid = threadIdx.x;
  const int lane = tid & 31;
  const int brow0 = blockIdx.x * 128;
  const int bcol0 = blockIdx.y * 128;

  v8f acc[2][4] = {};
  block_gemm_lds(o16 + (size_t)brow0 * Cq, Cq, Wp + (size_t)bcol0 * Cq, Cq, Cq,
                 Asm, Bsm, tid, acc);

  const int mrow = ((tid >> 5) & 3) * 32;
  const int ncol = ((tid >> 5) >> 2) * 64;
  const int cb = lane & 15;
  const int rb = (lane >> 4) << 3;
#pragma unroll
  for (int rt = 0; rt < 2; ++rt)
#pragma unroll
    for (int nt = 0; nt < 4; ++nt)
#pragma unroll
      for (int r = 0; r < 8; ++r) {
        const int row = brow0 + mrow + rt * 16 + rb + r;   // rolled domain
        const int col = bcol0 + ncol + nt * 16 + cb;
        const int b = row >> 12;
        const int p = row & (Sq - 1);
        const int s = (p + SHIFTq) & (Sq - 1);             // roll(+SHIFT)
        const size_t idx = (((size_t)(b << 12) + s)) * Cq + col;
        x1[idx] = x[idx] + acc[rt][nt][r] + bias[col];
      }
}

// ---------------- FC1 GEMM + exact GELU -> fp16 act -------------------------
__global__ __launch_bounds__(256) void fc1_kernel(const _Float16* __restrict__ h2,
                                                  const _Float16* __restrict__ W1,
                                                  const float* __restrict__ bias,
                                                  _Float16* __restrict__ act) {
  __shared__ alignas(16) _Float16 Asm[2 * 4096];
  __shared__ alignas(16) _Float16 Bsm[2 * 4096];
  const int tid = threadIdx.x;
  const int lane = tid & 31;
  const int brow0 = blockIdx.x * 128;
  const int bcol0 = blockIdx.y * 128;

  v8f acc[2][4] = {};
  block_gemm_lds(h2 + (size_t)brow0 * Cq, Cq, W1 + (size_t)bcol0 * Cq, Cq, Cq,
                 Asm, Bsm, tid, acc);

  const int mrow = ((tid >> 5) & 3) * 32;
  const int ncol = ((tid >> 5) >> 2) * 64;
  const int cb = lane & 15;
  const int rb = (lane >> 4) << 3;
#pragma unroll
  for (int rt = 0; rt < 2; ++rt)
#pragma unroll
    for (int nt = 0; nt < 4; ++nt)
#pragma unroll
      for (int r = 0; r < 8; ++r) {
        const int row = brow0 + mrow + rt * 16 + rb + r;
        const int col = bcol0 + ncol + nt * 16 + cb;
        float v = acc[rt][nt][r] + bias[col];
        float g = 0.5f * v * (1.0f + erff(v * 0.70710678118654752f));
        act[(size_t)row * (4 * Cq) + col] = (_Float16)g;
      }
}

// ---------------- FC2 GEMM + residual -> out (fp32) -------------------------
__global__ __launch_bounds__(256) void fc2_kernel(const _Float16* __restrict__ act,
                                                  const _Float16* __restrict__ W2,
                                                  const float* __restrict__ bias,
                                                  const float* __restrict__ x1,
                                                  float* __restrict__ out) {
  __shared__ alignas(16) _Float16 Asm[2 * 4096];
  __shared__ alignas(16) _Float16 Bsm[2 * 4096];
  const int tid = threadIdx.x;
  const int lane = tid & 31;
  const int brow0 = blockIdx.x * 128;
  const int bcol0 = blockIdx.y * 128;

  v8f acc[2][4] = {};
  block_gemm_lds(act + (size_t)brow0 * (4 * Cq), 4 * Cq,
                 W2 + (size_t)bcol0 * (4 * Cq), 4 * Cq, 4 * Cq,
                 Asm, Bsm, tid, acc);

  const int mrow = ((tid >> 5) & 3) * 32;
  const int ncol = ((tid >> 5) >> 2) * 64;
  const int cb = lane & 15;
  const int rb = (lane >> 4) << 3;
#pragma unroll
  for (int rt = 0; rt < 2; ++rt)
#pragma unroll
    for (int nt = 0; nt < 4; ++nt)
#pragma unroll
      for (int r = 0; r < 8; ++r) {
        const int row = brow0 + mrow + rt * 16 + rb + r;
        const int col = bcol0 + ncol + nt * 16 + cb;
        const size_t idx = (size_t)row * Cq + col;
        out[idx] = x1[idx] + acc[rt][nt][r] + bias[col];
      }
}

// ---------------- workspace layout (bytes) ----------------
static constexpr size_t OFF_QKVW = 0;
static constexpr size_t OFF_PROJW = OFF_QKVW + 1572864;
static constexpr size_t OFF_FC1W = OFF_PROJW + 524288;
static constexpr size_t OFF_FC2W = OFF_FC1W + 2097152;
static constexpr size_t OFF_H16  = OFF_FC2W + 2097152;
static constexpr size_t OFF_SPAN = OFF_H16 + 33554432;
static constexpr size_t OFF_Q16  = OFF_SPAN;
static constexpr size_t OFF_K16  = OFF_SPAN + 33554432;
static constexpr size_t OFF_V16  = OFF_SPAN + 67108864;
static constexpr size_t OFF_O16  = OFF_SPAN + 100663296;
static constexpr size_t OFF_ACT  = OFF_SPAN;                 // reuse q/k/v/o span
static constexpr size_t OFF_X1   = OFF_SPAN + 134217728;

extern "C" void kernel_launch(void* const* d_in, const int* in_sizes, int n_in,
                              void* d_out, int out_size, void* d_ws, size_t ws_size,
                              hipStream_t stream) {
  (void)in_sizes; (void)n_in; (void)out_size; (void)ws_size;
  const float* x       = (const float*)d_in[0];
  const float* n1g     = (const float*)d_in[1];
  const float* n1b     = (const float*)d_in[2];
  const float* qkv_w   = (const float*)d_in[3];
  const float* qkv_b   = (const float*)d_in[4];
  const float* rel_bias= (const float*)d_in[5];
  const float* proj_w  = (const float*)d_in[6];
  const float* proj_b  = (const float*)d_in[7];
  const float* n2g     = (const float*)d_in[8];
  const float* n2b     = (const float*)d_in[9];
  const float* fc1_w   = (const float*)d_in[10];
  const float* fc1_b   = (const float*)d_in[11];
  const float* fc2_w   = (const float*)d_in[12];
  const float* fc2_b   = (const float*)d_in[13];
  float* out = (float*)d_out;

  char* ws = (char*)d_ws;
  _Float16* qkvw16 = (_Float16*)(ws + OFF_QKVW);
  _Float16* projw16= (_Float16*)(ws + OFF_PROJW);
  _Float16* fc1w16 = (_Float16*)(ws + OFF_FC1W);
  _Float16* fc2w16 = (_Float16*)(ws + OFF_FC2W);
  _Float16* hN     = (_Float16*)(ws + OFF_H16);
  _Float16* q16    = (_Float16*)(ws + OFF_Q16);
  _Float16* k16    = (_Float16*)(ws + OFF_K16);
  _Float16* v16    = (_Float16*)(ws + OFF_V16);
  _Float16* o16    = (_Float16*)(ws + OFF_O16);
  _Float16* act16  = (_Float16*)(ws + OFF_ACT);
  float*    x1     = (float*)(ws + OFF_X1);

  // 1) weights fp32 -> fp16
  f32_to_f16_kernel<<<(786432 + 255) / 256, 256, 0, stream>>>(qkv_w, qkvw16, 786432);
  f32_to_f16_kernel<<<(262144 + 255) / 256, 256, 0, stream>>>(proj_w, projw16, 262144);
  f32_to_f16_kernel<<<(1048576 + 255) / 256, 256, 0, stream>>>(fc1_w, fc1w16, 1048576);
  f32_to_f16_kernel<<<(1048576 + 255) / 256, 256, 0, stream>>>(fc2_w, fc2w16, 1048576);

  // 2) LN1 + roll(-SHIFT) -> hN (fp16)
  ln_kernel<<<ROWS / 8, 256, 0, stream>>>(x, n1g, n1b, hN, SHIFTq);

  // 3) QKV GEMM
  { dim3 g(ROWS / 128, 1536 / 128);
    qkv_kernel<<<g, 256, 0, stream>>>(hN, qkvw16, qkv_b, q16, k16, v16); }

  // 4) windowed attention
  attn_kernel<<<Bq * NWq * Hq, 128, 0, stream>>>(q16, k16, v16, rel_bias, o16);

  // 5) proj GEMM + roll(+SHIFT) + residual -> x1
  { dim3 g(ROWS / 128, Cq / 128);
    proj_kernel<<<g, 256, 0, stream>>>(o16, projw16, proj_b, x, x1); }

  // 6) LN2 -> h2 (fp16), reuses hN buffer
  ln_kernel<<<ROWS / 8, 256, 0, stream>>>(x1, n2g, n2b, hN, 0);

  // 7) FC1 GEMM + GELU -> act16
  { dim3 g(ROWS / 128, (4 * Cq) / 128);
    fc1_kernel<<<g, 256, 0, stream>>>(hN, fc1w16, fc1_b, act16); }

  // 8) FC2 GEMM + residual -> out
  { dim3 g(ROWS / 128, Cq / 128);
    fc2_kernel<<<g, 256, 0, stream>>>(act16, fc2w16, fc2_b, x1, out); }
}
